// GNNStructEncoder_61409442398748
// MI455X (gfx1250) — compile-verified, matrix operand order checked by reading, not run on hardware
//
#include <hip/hip_runtime.h>
#include <stdint.h>

#define NN 20000
#define HH 128
#define DEGN 16
#define SS 5
#define KK 7
#define RR 3
#define WPAD (HH + 8)   // +16B row pad: keeps 16B alignment, breaks bank conflicts

typedef __attribute__((ext_vector_type(16))) __bf16 v16bf;
typedef __attribute__((ext_vector_type(8)))  __bf16 v8bf;
typedef __attribute__((ext_vector_type(8)))  float  v8f;
typedef __attribute__((ext_vector_type(4)))  uint32_t v4u;
typedef __attribute__((ext_vector_type(8)))  int    v8i;
typedef __attribute__((ext_vector_type(4)))  int    v4i;

#if defined(__has_builtin)
#if __has_builtin(__builtin_amdgcn_tensor_load_to_lds) && __has_builtin(__builtin_amdgcn_s_wait_tensorcnt)
#define HAVE_TDM 1
#endif
#endif

// ---------- helpers ----------
__device__ inline __bf16 f2bf(float f) {
  union { float f; uint32_t u; } v; v.f = f;
  uint32_t r = v.u + 0x7FFFu + ((v.u >> 16) & 1u);   // round-to-nearest-even
  uint16_t hs = (uint16_t)(r >> 16);
  return __builtin_bit_cast(__bf16, hs);
}

__device__ inline uint64_t splitmix64(uint64_t x) {
  x += 0x9E3779B97F4A7C15ull;
  x = (x ^ (x >> 30)) * 0xBF58476D1CE4E5B9ull;
  x = (x ^ (x >> 27)) * 0x94D049BB133111EBull;
  return x ^ (x >> 31);
}
__device__ inline float u01(uint64_t h) {
  return (float)(h >> 40) * (1.0f / 16777216.0f) + 1e-9f;   // (0,1)
}
__device__ inline float rnorm(uint64_t ctr) {
  float u1 = u01(splitmix64(ctr * 2ull + 0ull));
  float u2 = u01(splitmix64(ctr * 2ull + 1ull));
  return __builtin_sqrtf(-2.0f * __logf(u1)) * __cosf(6.2831853f * u2);
}
__device__ inline float wred(float v) {
  #pragma unroll
  for (int m = 16; m > 0; m >>= 1) v += __shfl_xor(v, m, 32);
  return v;
}

// WMMA bf16 A/B fragment (16x32 / 32x16, wave32): lane L holds row/col = L%16,
// K set = {hi8..hi8+7} u {hi8+16..hi8+23} within the kk window, where hi8=(L>=16)*8.
// With K contiguous in LDS this is exactly two ds_load_b128 per fragment.
__device__ inline v16bf frag16(const __bf16* rowbase, int kk, int hi8) {
  v8bf lo = *(const v8bf*)(rowbase + kk + hi8);
  v8bf hi = *(const v8bf*)(rowbase + kk + 16 + hi8);
  return __builtin_shufflevector(lo, hi, 0,1,2,3,4,5,6,7,8,9,10,11,12,13,14,15);
}

// ---------- kernel 0: zero the loss accumulator ----------
__global__ void k_zero(float* out) {
  if (threadIdx.x == 0 && blockIdx.x == 0) out[0] = 0.0f;
}

// ---------- kernel 1: degree loss (dd MLP is dead: softmax over axis of size 1 == 1) ----------
__global__ __launch_bounds__(256) void k_deg(const float* __restrict__ deg,
                                             float* __restrict__ out) {
  float a = 0.0f;
  for (int i = blockIdx.x * 256 + threadIdx.x; i < NN; i += gridDim.x * 256) {
    float d = 1.0f - deg[i];
    a += d * d;
  }
  a = wred(a);
  if ((threadIdx.x & 31) == 0) atomicAdd(out, a * (10.0f / (float)NN));
}

// ---------- kernel 2: fused gather + mlp1 (2 layers) + clf logits, bf16 WMMA ----------
__global__ __launch_bounds__(256) void k_gij(
    const float* __restrict__ h, const int* __restrict__ nbrs,
    const float* __restrict__ W1, const float* __restrict__ b1,
    const float* __restrict__ W2, const float* __restrict__ b2,
    const float* __restrict__ Wc, const float* __restrict__ bc,
    float* __restrict__ gij_out, float* __restrict__ logits_out) {
  __shared__ __bf16 sA [16][WPAD];   // A tile, K-contiguous rows (agg -> relu(out1) -> gij)
  __shared__ __bf16 sWt[HH][WPAD];   // W transposed: sWt[n][k], K-contiguous per output col
  __shared__ float  sB[HH];          // bias

  const int tid  = threadIdx.x;
  const int lane = tid & 31;
  const int wv   = tid >> 5;         // 8 waves; wave wv owns N-tile [wv*16, wv*16+16)
  const int node0 = blockIdx.x * 16;
  const int rowA = lane & 15;
  const int hi8  = (lane >> 4) << 3;
  const int n0   = wv * 16;
  const int col  = n0 + rowA;

  // ---- phase 0: gather-sum 17 rows per node -> bf16 A tile; stage W1^T, b1
  {
    const int m  = tid >> 4;              // node within tile
    const int c0 = (tid & 15) * 8;        // 8 contiguous columns
    const int node = node0 + m;
    float acc[8];
    const float* hr = h + (size_t)node * HH + c0;
    #pragma unroll
    for (int c = 0; c < 8; ++c) acc[c] = hr[c];
    for (int d = 0; d < DEGN; ++d) {
      const int nb = nbrs[node * DEGN + d];
      const float* rr = h + (size_t)nb * HH + c0;
      #pragma unroll
      for (int c = 0; c < 8; ++c) acc[c] += rr[c];
    }
    #pragma unroll
    for (int c = 0; c < 8; ++c) sA[m][c0 + c] = f2bf(acc[c]);
  }
  for (int i = tid; i < HH * HH; i += 256) {       // coalesced global read, transposed LDS write
    const int k = i >> 7, n = i & 127;
    sWt[n][k] = f2bf(W1[i]);
  }
  if (tid < HH) sB[tid] = b1[tid];
  __builtin_prefetch(W2, 0, 0);                    // warm next layer's weights
  __syncthreads();

  // ---- phase 1: out1 = agg @ W1
  v8f d1 = {};
  #pragma unroll
  for (int kk = 0; kk < HH; kk += 32) {
    v16bf a = frag16(&sA [rowA][0], kk, hi8);
    v16bf b = frag16(&sWt[col ][0], kk, hi8);
    d1 = __builtin_amdgcn_wmma_f32_16x16x32_bf16(false, a, false, b, (short)0, d1, false, false);
  }
  const float bias1 = sB[col];
  __syncthreads();

  // ---- phase 2: relu(out1+b1) -> A tile; stage W2^T, b2
  #pragma unroll
  for (int r = 0; r < 8; ++r) {
    const float v = d1[r] + bias1;
    sA[r + hi8][col] = f2bf(v > 0.0f ? v : 0.0f);
  }
  for (int i = tid; i < HH * HH; i += 256) {
    const int k = i >> 7, n = i & 127;
    sWt[n][k] = f2bf(W2[i]);
  }
  if (tid < HH) sB[tid] = b2[tid];
  __builtin_prefetch(Wc, 0, 0);
  __syncthreads();

  // ---- phase 3: gij = out1 @ W2 + b2
  v8f d2 = {};
  #pragma unroll
  for (int kk = 0; kk < HH; kk += 32) {
    v16bf a = frag16(&sA [rowA][0], kk, hi8);
    v16bf b = frag16(&sWt[col ][0], kk, hi8);
    d2 = __builtin_amdgcn_wmma_f32_16x16x32_bf16(false, a, false, b, (short)0, d2, false, false);
  }
  const float bias2 = sB[col];
  __syncthreads();

  // ---- phase 4: write gij (fp32 output), keep bf16 copy; stage padded Wc^T, bc
  #pragma unroll
  for (int r = 0; r < 8; ++r) {
    const int row = r + hi8;
    const float g = d2[r] + bias2;
    gij_out[(size_t)(node0 + row) * HH + col] = g;
    sA[row][col] = f2bf(g);
  }
  for (int i = tid; i < HH * 16; i += 256) {
    const int k = i >> 4, n = i & 15;
    sWt[n][k] = f2bf(n < KK ? Wc[k * KK + n] : 0.0f);
  }
  if (tid < 16) sB[tid] = (tid < KK) ? bc[tid] : 0.0f;
  __syncthreads();

  // ---- phase 5: logits = gij @ Wc + bc (single 16-col tile, wave 0)
  if (wv == 0) {
    v8f d3 = {};
    #pragma unroll
    for (int kk = 0; kk < HH; kk += 32) {
      v16bf a = frag16(&sA [rowA][0], kk, hi8);
      v16bf b = frag16(&sWt[rowA][0], kk, hi8);
      d3 = __builtin_amdgcn_wmma_f32_16x16x32_bf16(false, a, false, b, (short)0, d3, false, false);
    }
    if (rowA < KK) {
      #pragma unroll
      for (int r = 0; r < 8; ++r)
        logits_out[(size_t)(node0 + r + hi8) * KK + rowA] = d3[r] + sB[rowA];
    }
  }
}

// ---------- kernel 3: fused RNG + var + dropout + einsum + cost + 120-perm min ----------
// One wave per (r,n). Never materializes the (R,N,S,K,H) tensors (saves ~5 GB of HBM traffic).
// gmean/glogsig (17.9 KB each, contiguous) staged into LDS via the Tensor Data Mover.
__global__ __launch_bounds__(256) void k_loss(
    const float* __restrict__ h, const int* __restrict__ nbrs,
    const float* __restrict__ logits, const float* __restrict__ gmean,
    const float* __restrict__ glogsig, const int* __restrict__ temp,
    float* __restrict__ out) {
  __shared__ float s_gm[SS * KK * HH];
  __shared__ float s_sg[SS * KK * HH];
  const int tid = threadIdx.x;

#if defined(HAVE_TDM)
  if (tid < 32) {                       // wave 0 issues two 1-D TDM descriptors
    // D# group1: data_size=4B (bits17:16=2); tensor_dim0=tile_dim0=4480; stride0=4480
    v8i g1;
    g1[0] = 0x20000;                    // wg_mask=0, data_size=2 (4 bytes)
    g1[1] = (int)((SS * KK * HH) << 16);        // tensor_dim0 lo16 << 16
    g1[2] = 0x10000;                    // tensor_dim0 hi16=0 | tensor_dim1=1
    g1[3] = (int)((SS * KK * HH) << 16);        // tile_dim0 << 16
    g1[4] = 0;                          // tile_dim1=0 (unused), tile_dim2=0
    g1[5] = SS * KK * HH;               // tensor_dim0_stride lo32
    g1[6] = 0;
    g1[7] = 0;
    v4i z4 = {};
    v8i z8 = {};
    {
      const uint64_t ga = (uint64_t)(uintptr_t)gmean;
      v4u g0;
      g0[0] = 1u;                                       // count=1, user mode
      g0[1] = (uint32_t)(uintptr_t)&s_gm[0];            // lds_addr (low 32 bits of flat LDS addr)
      g0[2] = (uint32_t)ga;
      g0[3] = (uint32_t)((ga >> 32) & 0x01FFFFFFu) | 0x80000000u;  // type=2 ("image")
      __builtin_amdgcn_tensor_load_to_lds(g0, g1, z4, z4, z8, 0);
    }
    {
      const uint64_t ga = (uint64_t)(uintptr_t)glogsig;
      v4u g0;
      g0[0] = 1u;
      g0[1] = (uint32_t)(uintptr_t)&s_sg[0];
      g0[2] = (uint32_t)ga;
      g0[3] = (uint32_t)((ga >> 32) & 0x01FFFFFFu) | 0x80000000u;
      __builtin_amdgcn_tensor_load_to_lds(g0, g1, z4, z4, z8, 0);
    }
    __builtin_amdgcn_s_wait_tensorcnt(0);
  }
  __syncthreads();
  for (int i = tid; i < SS * KK * HH; i += 256) s_sg[i] = __expf(s_sg[i]);
#else
  for (int i = tid; i < SS * KK * HH; i += 256) {
    s_gm[i] = gmean[i];
    s_sg[i] = __expf(glogsig[i]);
  }
#endif
  __syncthreads();

  const int lane = tid & 31;
  const int widx = blockIdx.x * 8 + (tid >> 5);   // [0, R*N), N-major for locality
  const int n = widx % NN;
  const int r = widx / NN;
  const float invtemp = 1.0f / (float)(*temp);

  // zij = softmax((logits + gumbel)/temp) over K (redundant per lane, K=7)
  float zij[KK];
  {
    float mx = -3.4e38f;
    const uint64_t gbase = 0xA5A5A5A500000000ull + ((uint64_t)(r * NN + n)) * KK;
    #pragma unroll
    for (int k = 0; k < KK; ++k) {
      float u = u01(splitmix64(gbase + (uint64_t)k));
      float gum = -__logf(-__logf(u + 1e-20f) + 1e-20f);
      zij[k] = (logits[n * KK + k] + gum) * invtemp;
      mx = fmaxf(mx, zij[k]);
    }
    float sum = 0.0f;
    #pragma unroll
    for (int k = 0; k < KK; ++k) { zij[k] = __expf(zij[k] - mx); sum += zij[k]; }
    const float inv = 1.0f / sum;
    #pragma unroll
    for (int k = 0; k < KK; ++k) zij[k] *= inv;
  }

  // gen[r,n,s,h] = sum_k zij[k] * dropout(gmean + exp(glogsig)*N(0,1)) — RNG in registers
  float gen[SS][4];
  const uint64_t nbase = ((uint64_t)widx) * (uint64_t)(SS * KK * HH);
  #pragma unroll
  for (int s = 0; s < SS; ++s) {
    #pragma unroll
    for (int c = 0; c < 4; ++c) gen[s][c] = 0.0f;
    for (int k = 0; k < KK; ++k) {
      const float zk = zij[k];
      const int o = (s * KK + k) * HH;
      #pragma unroll
      for (int c = 0; c < 4; ++c) {
        const int hc = lane + 32 * c;
        const uint64_t ctr = nbase + (uint64_t)(o + hc);
        float v = s_gm[o + hc] + s_sg[o + hc] * rnorm(ctr);
        const float uk = u01(splitmix64(ctr ^ 0xD1B54A32D192ED03ull));
        v = (uk < 0.8f) ? v * 1.25f : 0.0f;
        gen[s][c] = fmaf(zk, v, gen[s][c]);
      }
    }
  }

  // targets: first S neighbors' feature rows
  float tg[SS][4];
  #pragma unroll
  for (int s = 0; s < SS; ++s) {
    const int row = nbrs[n * DEGN + s];
    #pragma unroll
    for (int c = 0; c < 4; ++c) tg[s][c] = h[(size_t)row * HH + lane + 32 * c];
  }

  // cost[i][j] = |gen_i|^2/H + |tgt_j|^2/H - 2 <gen_i,tgt_j>/H  (butterfly reductions)
  float g2[SS], t2[SS], C[SS][SS];
  #pragma unroll
  for (int i = 0; i < SS; ++i) {
    float a = 0.0f;
    #pragma unroll
    for (int c = 0; c < 4; ++c) a += gen[i][c] * gen[i][c];
    g2[i] = wred(a) * (1.0f / HH);
  }
  #pragma unroll
  for (int j = 0; j < SS; ++j) {
    float a = 0.0f;
    #pragma unroll
    for (int c = 0; c < 4; ++c) a += tg[j][c] * tg[j][c];
    t2[j] = wred(a) * (1.0f / HH);
  }
  #pragma unroll
  for (int i = 0; i < SS; ++i)
    #pragma unroll
    for (int j = 0; j < SS; ++j) {
      float a = 0.0f;
      #pragma unroll
      for (int c = 0; c < 4; ++c) a += gen[i][c] * tg[j][c];
      C[i][j] = g2[i] + t2[j] - 2.0f * wred(a) * (1.0f / HH);
    }

  // min over all 120 permutations of mean_i C[i][perm[i]]
  if (lane == 0) {
    float best = 3.4e38f;
    for (int a = 0; a < 5; ++a)
      for (int b = 0; b < 5; ++b) {
        if (b == a) continue;
        for (int c = 0; c < 5; ++c) {
          if (c == a || c == b) continue;
          for (int d = 0; d < 5; ++d) {
            if (d == a || d == b || d == c) continue;
            const int e = 10 - a - b - c - d;
            const float sum = C[0][a] + C[1][b] + C[2][c] + C[3][d] + C[4][e];
            best = fminf(best, sum);
          }
        }
      }
    atomicAdd(out, best * (0.2f / (float)RR));   // /S then mean over R of sum over N
  }
}

// ---------- launcher ----------
// Input order assumed (setup_inputs insertion order, params flattened in insertion order):
// 0:h 1:nbrs 2:degrees 3:W1 4:b1 5:W2 6:b2 7:Wclf 8:bclf 9..16:dd(dead) 17,18:dd2(dead)
// 19:gmean 20:glogsig 21:temp
extern "C" void kernel_launch(void* const* d_in, const int* in_sizes, int n_in,
                              void* d_out, int out_size, void* d_ws, size_t ws_size,
                              hipStream_t stream) {
  const float* h        = (const float*)d_in[0];
  const int*   nbrs     = (const int*)  d_in[1];
  const float* degrees  = (const float*)d_in[2];
  const float* W1       = (const float*)d_in[3];
  const float* b1       = (const float*)d_in[4];
  const float* W2       = (const float*)d_in[5];
  const float* b2       = (const float*)d_in[6];
  const float* Wc       = (const float*)d_in[7];
  const float* bc       = (const float*)d_in[8];
  const float* gmean    = (const float*)d_in[19];
  const float* glogsig  = (const float*)d_in[20];
  const int*   temp     = (const int*)  d_in[21];

  float* out    = (float*)d_out;       // [0] = loss, [1..] = gij (N*H)
  float* gij    = out + 1;
  float* logits = (float*)d_ws;        // N*K fp32 scratch (560 KB)

  k_zero<<<1, 32, 0, stream>>>(out);
  k_gij <<<NN / 16, 256, 0, stream>>>(h, nbrs, W1, b1, W2, b2, Wc, bc, gij, logits);
  k_deg <<<32, 256, 0, stream>>>(degrees, out);
  k_loss<<<(RR * NN) / 8, 256, 0, stream>>>(h, nbrs, logits, gmean, glogsig, temp, out);
}
